// MultiHeadedAttention_38852274160163
// MI455X (gfx1250) — matrix operand order repaired and checked by reading, not compiled, not run on hardware
//
#include <hip/hip_runtime.h>
#include <hip/hip_bf16.h>

#define B_  2
#define S_  2048
#define D_  1024
#define H_  16
#define HD_ 64

typedef __attribute__((ext_vector_type(16))) _Float16 v16h;
typedef __attribute__((ext_vector_type(8)))  _Float16 v8h;
typedef __attribute__((ext_vector_type(4)))  _Float16 v4h;
typedef __attribute__((ext_vector_type(8)))  float    v8f;
typedef __attribute__((ext_vector_type(4)))  unsigned int u32x4;
typedef __attribute__((ext_vector_type(8)))  int      i32x8;

// ---------------- WMMA helpers ----------------

__device__ __forceinline__ v8f wmma_f16(v16h a, v16h b, v8f c) {
  // D = A(16x32 f16) * B(32x16 f16) + C(16x16 f32)
  return __builtin_amdgcn_wmma_f32_16x16x32_f16(
      /*neg_a=*/false, a, /*neg_b=*/false, b,
      /*c_mod=*/(short)0, c, /*reuse_a=*/false, /*reuse_b=*/false);
}

__device__ __forceinline__ v16h pack2(v8h lo, v8h hi) {
  v16h r;
#pragma unroll
  for (int i = 0; i < 8; ++i) { r[i] = lo[i]; r[i + 8] = hi[i]; }
  return r;
}

// A fragment: 16x32 f16 tile from row-major [M][K] source, leading dim `ld`.
// Lane l holds row M=l&15; lanes 0-15: K 0..7 & 16..23, lanes 16-31: K 8..15 & 24..31.
__device__ __forceinline__ v16h load_frag_a(const _Float16* p, int ld, int lane) {
  const _Float16* r = p + (size_t)(lane & 15) * ld + ((lane >> 4) << 3);
  return pack2(*(const v8h*)r, *(const v8h*)(r + 16));
}

// B fragment: 32x16 f16 tile, source stored transposed [N][K] row-major.
// Lane l holds column N=l&15; lanes 0-15: K 0..7 & 16..23, lanes 16-31: K 8..15 & 24..31.
__device__ __forceinline__ v16h load_frag_b(const _Float16* p, int ld, int lane) {
  const _Float16* r = p + (size_t)(lane & 15) * ld + ((lane >> 4) << 3);
  return pack2(*(const v8h*)r, *(const v8h*)(r + 16));
}

// xor-shuffle within 16-lane groups via ds_swizzle (group-of-32 mode, and_mask=0x1f)
template <int XM>
__device__ __forceinline__ float swz_xor(float v) {
  int i = __builtin_amdgcn_ds_swizzle(__float_as_int(v), (XM << 10) | 0x1f);
  return __int_as_float(i);
}

// ---------------- Tensor Data Mover: 2D tile -> LDS ----------------
// D# per CDNA5 ISA (2 groups => tensors up to 2D).
// Generic LDS pointers carry the LDS byte offset in addr[31:0] (aperture rule).

__device__ __forceinline__ unsigned int lds_off(const void* p) {
  return (unsigned int)(unsigned long long)p;
}

__device__ __forceinline__ void tdm_load_2d(unsigned int lds_byte, const void* gptr,
                                            unsigned int tile_k, unsigned int tile_rows,
                                            unsigned int stride_elems) {
  unsigned long long ga = (unsigned long long)gptr;
  u32x4 g0;
  g0[0] = 1u;                                              // count=1 (valid user D#)
  g0[1] = lds_byte;                                        // lds_addr (bytes)
  g0[2] = (unsigned int)ga;                                // global_addr[31:0]
  g0[3] = ((unsigned int)(ga >> 32) & 0x01FFFFFFu) | (2u << 30);  // addr[56:32], type=2
  const unsigned int td0 = 1u << 20, td1 = 1u << 20;       // generous tensor dims (no OOB)
  i32x8 g1;
  g1[0] = (int)(1u << 16);                                 // wg_mask=0, data_size=1 (2 bytes)
  g1[1] = (int)((td0 & 0xFFFFu) << 16);                    // tensor_dim0[15:0]
  g1[2] = (int)((td0 >> 16) | ((td1 & 0xFFFFu) << 16));    // tensor_dim0[31:16], dim1[15:0]
  g1[3] = (int)((td1 >> 16) | (tile_k << 16));             // tensor_dim1[31:16], tile_dim0
  g1[4] = (int)(tile_rows & 0xFFFFu);                      // tile_dim1 (tile_dim2=0)
  g1[5] = (int)stride_elems;                               // tensor_dim0_stride[31:0]
  g1[6] = 0;                                               // stride0[47:32], stride1[15:0]
  g1[7] = 0;                                               // stride1[47:16]
  asm volatile("tensor_load_to_lds %0, %1" :: "s"(g0), "s"(g1) : "memory");
}

// ---------------- Kernel 1: f32 -> f16 convert ----------------

__global__ void mha_cvt_f16(const float4* __restrict__ src, v4h* __restrict__ dst, int n4) {
  int i = blockIdx.x * blockDim.x + threadIdx.x;
  if (i < n4) {
    float4 v = src[i];
    v4h o;
    o[0] = (_Float16)v.x; o[1] = (_Float16)v.y;
    o[2] = (_Float16)v.z; o[3] = (_Float16)v.w;
    dst[i] = o;
  }
}

// ---------------- Kernel 2: weight transpose + convert (W[k][n] -> Wt[n][k] f16) ----

__global__ __launch_bounds__(256) void mha_wtrans(const float* __restrict__ W,
                                                  _Float16* __restrict__ Wt) {
  __shared__ float tile[32][33];
  int bx = blockIdx.x * 32;  // n base
  int by = blockIdx.y * 32;  // k base
  int tx = threadIdx.x, ty = threadIdx.y;  // 32 x 8
#pragma unroll
  for (int i = ty; i < 32; i += 8)
    tile[i][tx] = W[(size_t)(by + i) * D_ + bx + tx];
  __syncthreads();
#pragma unroll
  for (int i = ty; i < 32; i += 8)
    Wt[(size_t)(bx + i) * D_ + by + tx] = (_Float16)tile[tx][i];
}

// ---------------- TDM-fed, double-buffered LDS GEMM core ----------------
// Block = 256 threads (8 waves) computes a 128(M) x 128(N) tile, K=1024.
// Wave (wm=wave>>1, wn=wave&1) computes 32x64; acc[2][4] of 16x16 f32.

__device__ __forceinline__ void gemm_mainloop_tdm(
    const _Float16* __restrict__ X,   // [M][1024] row-major (A source)
    const _Float16* __restrict__ Wt,  // [N][1024] row-major (B^T source)
    _Float16* ldsA, _Float16* ldsB,   // [2][128*64] halves each
    int m_blk, int n_blk, int wave, int lane, v8f acc[2][4]) {
  const int wave_m = wave >> 1;  // 0..3 -> M offset *32
  const int wave_n = wave & 1;   // 0..1 -> N offset *64
  if (wave == 0) {
    tdm_load_2d(lds_off(ldsA), X + (size_t)m_blk * D_, 64, 128, D_);
    tdm_load_2d(lds_off(ldsB), Wt + (size_t)n_blk * D_, 64, 128, D_);
    __builtin_amdgcn_s_wait_tensorcnt(0);
  }
  __syncthreads();
  for (int ks = 0; ks < D_ / 64; ++ks) {
    const int buf = ks & 1;
    if (wave == 0 && ks + 1 < D_ / 64) {
      tdm_load_2d(lds_off(ldsA + (buf ^ 1) * 128 * 64),
                  X + (size_t)m_blk * D_ + (ks + 1) * 64, 64, 128, D_);
      tdm_load_2d(lds_off(ldsB + (buf ^ 1) * 128 * 64),
                  Wt + (size_t)n_blk * D_ + (ks + 1) * 64, 64, 128, D_);
    }
    const _Float16* A  = ldsA + buf * 128 * 64;
    const _Float16* Bp = ldsB + buf * 128 * 64;
#pragma unroll
    for (int kk = 0; kk < 64; kk += 32) {
      v16h a0 = load_frag_a(A + (size_t)(wave_m * 32 + 0) * 64 + kk, 64, lane);
      v16h a1 = load_frag_a(A + (size_t)(wave_m * 32 + 16) * 64 + kk, 64, lane);
#pragma unroll
      for (int t = 0; t < 4; ++t) {
        v16h b = load_frag_b(Bp + (size_t)(wave_n * 64 + t * 16) * 64 + kk, 64, lane);
        acc[0][t] = wmma_f16(a0, b, acc[0][t]);
        acc[1][t] = wmma_f16(a1, b, acc[1][t]);
      }
    }
    if (wave == 0) __builtin_amdgcn_s_wait_tensorcnt(0);
    __syncthreads();
  }
}

// ---------------- Kernel 3: fused QKV projection GEMM ----------------
// grid: (1024/128, 4096/128, 3), block 256.
// z=0 -> Q head-split [b,h,s,d]; z=1 -> K head-split; z=2 -> V transposed [b,h,d,s].

__global__ __launch_bounds__(256) void mha_proj_gemm(
    const _Float16* __restrict__ xq, const _Float16* __restrict__ xk,
    const _Float16* __restrict__ xv,
    const _Float16* __restrict__ WqT, const _Float16* __restrict__ WkT,
    const _Float16* __restrict__ WvT,
    const float* __restrict__ bq, const float* __restrict__ bk,
    const float* __restrict__ bv,
    _Float16* __restrict__ Qh, _Float16* __restrict__ Kh,
    _Float16* __restrict__ Vt) {
  __shared__ __align__(16) _Float16 ldsA[2 * 128 * 64];
  __shared__ __align__(16) _Float16 ldsB[2 * 128 * 64];
  const int z = blockIdx.z;
  const _Float16* X  = (z == 0) ? xq  : (z == 1) ? xk  : xv;
  const _Float16* Wt = (z == 0) ? WqT : (z == 1) ? WkT : WvT;
  const float* bias  = (z == 0) ? bq  : (z == 1) ? bk  : bv;

  const int wave = threadIdx.x >> 5;
  const int lane = threadIdx.x & 31;
  const int m_blk = blockIdx.y * 128;
  const int n_blk = blockIdx.x * 128;

  v8f acc[2][4] = {};
  gemm_mainloop_tdm(X, Wt, ldsA, ldsB, m_blk, n_blk, wave, lane, acc);

  const int wave_m = wave >> 1, wave_n = wave & 1;
  const int half = lane >> 4, nn = lane & 15;
#pragma unroll
  for (int sub = 0; sub < 2; ++sub)
#pragma unroll
    for (int t = 0; t < 4; ++t)
#pragma unroll
      for (int j = 0; j < 8; ++j) {
        int m = m_blk + wave_m * 32 + sub * 16 + j + 8 * half;  // row in [B*S]
        int n = n_blk + wave_n * 64 + t * 16 + nn;              // col in [H*HD]
        float val = acc[sub][t][j] + bias[n];
        int bi = m >> 11, s = m & (S_ - 1);
        int h = n >> 6, d = n & (HD_ - 1);
        if (z == 0)
          Qh[(((size_t)bi * H_ + h) * S_ + s) * HD_ + d] = (_Float16)val;
        else if (z == 1)
          Kh[(((size_t)bi * H_ + h) * S_ + s) * HD_ + d] = (_Float16)val;
        else
          Vt[(((size_t)bi * H_ + h) * HD_ + d) * S_ + s] = (_Float16)val;
      }
}

// ---------------- Kernel 4: flash attention, 1 wave per (b,h,16-query tile) ----

__global__ __launch_bounds__(32) void mha_attn(
    const _Float16* __restrict__ Qh, const _Float16* __restrict__ Kh,
    const _Float16* __restrict__ Vt, const int* __restrict__ mask,
    _Float16* __restrict__ X16) {
  __shared__ __align__(16) _Float16 pslab[16 * 32];

  const int lane = threadIdx.x & 31;
  const int qb = blockIdx.x, h = blockIdx.y, bi = blockIdx.z;
  const int half = lane >> 4, nn = lane & 15;

  const _Float16* Q = Qh + ((size_t)bi * H_ + h) * S_ * HD_;
  const _Float16* K = Kh + ((size_t)bi * H_ + h) * S_ * HD_;
  const _Float16* V = Vt + ((size_t)bi * H_ + h) * HD_ * (size_t)S_;
  const int* M = mask + ((size_t)bi * S_ + qb * 16) * S_;

  v16h aq0 = load_frag_a(Q + (size_t)qb * 16 * HD_, HD_, lane);
  v16h aq1 = load_frag_a(Q + (size_t)qb * 16 * HD_ + 32, HD_, lane);

  float mrun[8], lrun[8];
#pragma unroll
  for (int j = 0; j < 8; ++j) { mrun[j] = -3.0e38f; lrun[j] = 0.0f; }
  v8f o[4] = {};

  for (int kc = 0; kc < S_; kc += 32) {
    v8f s0 = {}, s1 = {};
    {
      v16h b00 = load_frag_b(K + (size_t)(kc + 0) * HD_ + 0, HD_, lane);
      v16h b01 = load_frag_b(K + (size_t)(kc + 0) * HD_ + 32, HD_, lane);
      s0 = wmma_f16(aq0, b00, s0);
      s0 = wmma_f16(aq1, b01, s0);
      v16h b10 = load_frag_b(K + (size_t)(kc + 16) * HD_ + 0, HD_, lane);
      v16h b11 = load_frag_b(K + (size_t)(kc + 16) * HD_ + 32, HD_, lane);
      s1 = wmma_f16(aq0, b10, s1);
      s1 = wmma_f16(aq1, b11, s1);
    }
    float sv[2][8];
#pragma unroll
    for (int f = 0; f < 2; ++f)
#pragma unroll
      for (int j = 0; j < 8; ++j) {
        float s = (f ? s1[j] : s0[j]) * 0.125f;  // 1/sqrt(64)
        int qrow = j + 8 * half;
        int kcol = kc + f * 16 + nn;
        if (M[(size_t)qrow * S_ + kcol] == 0) s = -1.0e9f;
        sv[f][j] = s;
      }
    float mc[8];
#pragma unroll
    for (int j = 0; j < 8; ++j) mc[j] = fmaxf(sv[0][j], sv[1][j]);
#pragma unroll
    for (int j = 0; j < 8; ++j) {
      mc[j] = fmaxf(mc[j], swz_xor<1>(mc[j]));
      mc[j] = fmaxf(mc[j], swz_xor<2>(mc[j]));
      mc[j] = fmaxf(mc[j], swz_xor<4>(mc[j]));
      mc[j] = fmaxf(mc[j], swz_xor<8>(mc[j]));
    }
    float fac[8], lc[8];
#pragma unroll
    for (int j = 0; j < 8; ++j) {
      float mn = fmaxf(mrun[j], mc[j]);
      fac[j] = __expf(mrun[j] - mn);
      mrun[j] = mn;
      lc[j] = 0.0f;
    }
#pragma unroll
    for (int f = 0; f < 2; ++f)
#pragma unroll
      for (int j = 0; j < 8; ++j) {
        float p = __expf(sv[f][j] - mrun[j]);
        lc[j] += p;
        pslab[(j + 8 * half) * 32 + f * 16 + nn] = (_Float16)p;
      }
#pragma unroll
    for (int j = 0; j < 8; ++j) {
      lc[j] += swz_xor<1>(lc[j]);
      lc[j] += swz_xor<2>(lc[j]);
      lc[j] += swz_xor<4>(lc[j]);
      lc[j] += swz_xor<8>(lc[j]);
      lrun[j] = lrun[j] * fac[j] + lc[j];
    }
#pragma unroll
    for (int t = 0; t < 4; ++t)
#pragma unroll
      for (int j = 0; j < 8; ++j) o[t][j] *= fac[j];
    __syncthreads();
    v16h ap = load_frag_a(pslab, 32, lane);
#pragma unroll
    for (int t = 0; t < 4; ++t) {
      v16h bv = load_frag_b(V + (size_t)(t * 16) * S_ + kc, S_, lane);
      o[t] = wmma_f16(ap, bv, o[t]);
    }
    __syncthreads();
  }

#pragma unroll
  for (int t = 0; t < 4; ++t)
#pragma unroll
    for (int j = 0; j < 8; ++j) {
      int q = qb * 16 + j + 8 * half;
      float val = o[t][j] / lrun[j];
      X16[((size_t)bi * S_ + q) * D_ + h * HD_ + t * 16 + nn] = (_Float16)val;
    }
}

// ---------------- Kernel 5: output projection -> fp32 d_out ----------------

__global__ __launch_bounds__(256) void mha_out_gemm(
    const _Float16* __restrict__ X16, const _Float16* __restrict__ WoT,
    const float* __restrict__ bo, float* __restrict__ out) {
  __shared__ __align__(16) _Float16 ldsA[2 * 128 * 64];
  __shared__ __align__(16) _Float16 ldsB[2 * 128 * 64];
  const int wave = threadIdx.x >> 5;
  const int lane = threadIdx.x & 31;
  const int m_blk = blockIdx.y * 128;
  const int n_blk = blockIdx.x * 128;

  v8f acc[2][4] = {};
  gemm_mainloop_tdm(X16, WoT, ldsA, ldsB, m_blk, n_blk, wave, lane, acc);

  const int wave_m = wave >> 1, wave_n = wave & 1;
  const int half = lane >> 4, nn = lane & 15;
#pragma unroll
  for (int sub = 0; sub < 2; ++sub)
#pragma unroll
    for (int t = 0; t < 4; ++t)
#pragma unroll
      for (int j = 0; j < 8; ++j) {
        int m = m_blk + wave_m * 32 + sub * 16 + j + 8 * half;
        int n = n_blk + wave_n * 64 + t * 16 + nn;
        out[(size_t)m * D_ + n] = acc[sub][t][j] + bo[n];
      }
}

// ---------------- launch ----------------

extern "C" void kernel_launch(void* const* d_in, const int* in_sizes, int n_in,
                              void* d_out, int out_size, void* d_ws, size_t ws_size,
                              hipStream_t stream) {
  const float* query = (const float*)d_in[0];
  const float* key   = (const float*)d_in[1];
  const float* value = (const float*)d_in[2];
  const int*   mask  = (const int*)d_in[3];
  const float* Wq = (const float*)d_in[4];
  const float* bq = (const float*)d_in[5];
  const float* Wk = (const float*)d_in[6];
  const float* bk = (const float*)d_in[7];
  const float* Wv = (const float*)d_in[8];
  const float* bv = (const float*)d_in[9];
  const float* Wo = (const float*)d_in[10];
  const float* bo = (const float*)d_in[11];
  float* out = (float*)d_out;

  const size_t nx = (size_t)B_ * S_ * D_;  // 4,194,304
  char* w = (char*)d_ws;
  _Float16* xq16 = (_Float16*)w; w += nx * 2;
  _Float16* xk16 = (_Float16*)w; w += nx * 2;
  _Float16* xv16 = (_Float16*)w; w += nx * 2;
  _Float16* WqT  = (_Float16*)w; w += (size_t)D_ * D_ * 2;
  _Float16* WkT  = (_Float16*)w; w += (size_t)D_ * D_ * 2;
  _Float16* WvT  = (_Float16*)w; w += (size_t)D_ * D_ * 2;
  _Float16* WoT  = (_Float16*)w; w += (size_t)D_ * D_ * 2;
  _Float16* Qh   = (_Float16*)w; w += nx * 2;
  _Float16* Kh   = (_Float16*)w; w += nx * 2;
  _Float16* Vt   = (_Float16*)w; w += nx * 2;
  _Float16* X16  = (_Float16*)w; w += nx * 2;
  // total workspace: 64 MB

  {
    int n4 = (int)(nx / 4);
    dim3 g((n4 + 255) / 256), b(256);
    mha_cvt_f16<<<g, b, 0, stream>>>((const float4*)query, (v4h*)xq16, n4);
    mha_cvt_f16<<<g, b, 0, stream>>>((const float4*)key,   (v4h*)xk16, n4);
    mha_cvt_f16<<<g, b, 0, stream>>>((const float4*)value, (v4h*)xv16, n4);
  }
  {
    dim3 g(D_ / 32, D_ / 32), b(32, 8);
    mha_wtrans<<<g, b, 0, stream>>>(Wq, WqT);
    mha_wtrans<<<g, b, 0, stream>>>(Wk, WkT);
    mha_wtrans<<<g, b, 0, stream>>>(Wv, WvT);
    mha_wtrans<<<g, b, 0, stream>>>(Wo, WoT);
  }
  {
    dim3 g(D_ / 128, (B_ * S_) / 128, 3), b(256);
    mha_proj_gemm<<<g, b, 0, stream>>>(xq16, xk16, xv16, WqT, WkT, WvT,
                                       bq, bk, bv, Qh, Kh, Vt);
  }
  {
    dim3 g(S_ / 16, H_, B_), b(32);
    mha_attn<<<g, b, 0, stream>>>(Qh, Kh, Vt, mask, X16);
  }
  {
    dim3 g(D_ / 128, (B_ * S_) / 128), b(256);
    mha_out_gemm<<<g, b, 0, stream>>>(X16, WoT, bo, out);
  }
}
